// AttentionWithRoPE_16320875724862
// MI455X (gfx1250) — matrix operand order repaired and checked by reading, not compiled, not run on hardware
//
#include <hip/hip_runtime.h>
#include <stdint.h>

// Problem constants (from reference)
#define BN   2
#define SN   2048
#define HDIM 2048
#define NHQ  16
#define KVH  4
#define HD   128
#define GROUPS 4

typedef __attribute__((ext_vector_type(16))) __bf16 v16bf;
typedef __attribute__((ext_vector_type(8)))  float  v8f;

struct Frag32B { uint4 lo; uint4 hi; };

// Build a 16-element bf16 WMMA fragment from two 16-byte chunks.
__device__ __forceinline__ v16bf make_frag(const unsigned short* c0, const unsigned short* c1) {
  Frag32B f;
  f.lo = *reinterpret_cast<const uint4*>(c0);
  f.hi = *reinterpret_cast<const uint4*>(c1);
  return __builtin_bit_cast(v16bf, f);
}

__device__ __forceinline__ unsigned short f2bf(float x) {
  unsigned int u = __builtin_bit_cast(unsigned int, x);
  u += 0x7FFFu + ((u >> 16) & 1u);   // round-to-nearest-even
  return (unsigned short)(u >> 16);
}

#define WMMA_BF16(a, b, c) \
  __builtin_amdgcn_wmma_f32_16x16x32_bf16(false, (a), false, (b), (short)0, (c), false, false)

// ---------------------------------------------------------------- convert
__global__ void cvt_f32_bf16(const float* __restrict__ in,
                             unsigned short* __restrict__ out, int n) {
  int i = blockIdx.x * blockDim.x + threadIdx.x;
  if (i < n) out[i] = f2bf(in[i]);
}

// ---------------------------------------------------------------- GEMM  C[M,N] = A[M,K] * B[N,K]^T
// Block: 256 threads = 8 waves (2x4 wave grid). Wave tile 64x64 = 4x4 wmma.
// Block tile 128x256. 16 WMMAs per 8 fragment loads -> 32 flops/byte from L2.
__global__ __launch_bounds__(256)
void gemm_bf16_nt(const unsigned short* __restrict__ A,
                  const unsigned short* __restrict__ B,
                  float* __restrict__ C, int M, int N, int K) {
  const int lane = threadIdx.x & 31;
  const int wave = threadIdx.x >> 5;
  const int half = lane >> 4;
  const int lr   = lane & 15;
  const int wm   = wave >> 2;      // 0..1
  const int wn   = wave & 3;       // 0..3
  const int m0 = blockIdx.y * 128 + wm * 64;
  const int n0 = blockIdx.x * 256 + wn * 64;

  v8f acc[4][4];
#pragma unroll
  for (int i = 0; i < 4; ++i)
#pragma unroll
    for (int j = 0; j < 4; ++j) { v8f z = {}; acc[i][j] = z; }

  const unsigned short* ar[4];
  const unsigned short* br[4];
#pragma unroll
  for (int i = 0; i < 4; ++i) {
    ar[i] = A + (size_t)(m0 + 16 * i + lr) * K;
    br[i] = B + (size_t)(n0 + 16 * i + lr) * K;
  }

  for (int k0 = 0; k0 < K; k0 += 32) {
    const int ka = k0 + 8 * half;    // A: two 8-elt chunks at ka, ka+16
    const int kb = k0 + 8 * half;    // B: two 8-elt chunks at kb, kb+16
    __builtin_prefetch(ar[0] + k0 + 256, 0, 3);
    __builtin_prefetch(br[0] + k0 + 256, 0, 3);
    v16bf af[4], bf[4];
#pragma unroll
    for (int i = 0; i < 4; ++i) {
      af[i] = make_frag(ar[i] + ka, ar[i] + ka + 16);
      bf[i] = make_frag(br[i] + kb, br[i] + kb + 16);
    }
#pragma unroll
    for (int i = 0; i < 4; ++i)
#pragma unroll
      for (int j = 0; j < 4; ++j)
        acc[i][j] = WMMA_BF16(af[i], bf[j], acc[i][j]);
  }

#pragma unroll
  for (int i = 0; i < 4; ++i)
#pragma unroll
    for (int j = 0; j < 4; ++j)
#pragma unroll
      for (int v = 0; v < 8; ++v) {
        const int r = m0 + 16 * i + v + 8 * half;
        C[(size_t)r * N + n0 + 16 * j + lr] = acc[i][j][v];
      }
}

// ---------------------------------------------------------------- RoPE + layout
// in: [B,S,nh*HD] f32.  doRope: apply rotary (fp32).
// transposeDS=0 -> out [B,nh,S,HD]; transposeDS=1 -> out [B,nh,HD,S].
__global__ void rope_transpose(const float* __restrict__ in,
                               const float* __restrict__ cosp,
                               const float* __restrict__ sinp,
                               unsigned short* __restrict__ out,
                               int nh, int doRope, int transposeDS) {
  int i = blockIdx.x * blockDim.x + threadIdx.x;
  int total = BN * SN * nh * HD;
  if (i >= total) return;
  int d = i % HD; int t = i / HD;
  int h = t % nh; t /= nh;
  int s = t % SN; int b = t / SN;
  const float* src = in + (((size_t)(b * SN + s)) * nh + h) * HD;
  float x = src[d];
  float val = x;
  if (doRope) {
    float c  = cosp[s * HD + d];
    float sn = sinp[s * HD + d];
    float other = (d < HD / 2) ? -src[d + HD / 2] : src[d - HD / 2];
    val = x * c + other * sn;
  }
  size_t o = transposeDS
                 ? ((((size_t)b * nh + h) * HD + d) * SN + s)
                 : ((((size_t)b * nh + h) * SN + s) * HD + d);
  out[o] = f2bf(val);
}

// ---------------------------------------------------------------- flash attention
// One wave handles 16 query rows of one (b, kv-head, group). Causal, online softmax.
// Qb [B,NHQ,S,HD] bf16; Kb [B,KVH,S,HD] bf16; Vt [B,KVH,HD,S] bf16; AO [B,S,NHQ*HD] bf16.
__global__ __launch_bounds__(256)
void flash_attn(const unsigned short* __restrict__ Qb,
                const unsigned short* __restrict__ Kb,
                const unsigned short* __restrict__ Vt,
                unsigned short* __restrict__ AO) {
  __shared__ unsigned short Plds[8 * 16 * 40];   // per-wave 16x32 P tile, padded rows
  const int lane = threadIdx.x & 31;
  const int wave = threadIdx.x >> 5;
  const int half = lane >> 4;
  const int lr   = lane & 15;
  unsigned short* Pl = &Plds[wave * 640];

  int g = blockIdx.x * 8 + wave;
  const int qblocks = SN / 16;
  const int qb  = g % qblocks; g /= qblocks;
  const int grp = g % GROUPS;  g /= GROUPS;
  const int kv  = g % KVH;
  const int b   = g / KVH;
  const int h   = kv * GROUPS + grp;
  const int q0  = qb * 16;

  const unsigned short* Qh = Qb + ((size_t)b * NHQ + h)  * SN * HD;
  const unsigned short* Kh = Kb + ((size_t)b * KVH + kv) * SN * HD;
  const unsigned short* Vh = Vt + ((size_t)b * KVH + kv) * HD * SN;

  // Q fragments (A layout, rows = queries, 4 chunks of 32 dims)
  v16bf qf[4];
  const unsigned short* qrow = Qh + (size_t)(q0 + lr) * HD;
#pragma unroll
  for (int j = 0; j < 4; ++j) {
    const int ka = 32 * j + 8 * half;
    qf[j] = make_frag(qrow + ka, qrow + ka + 16);
  }

  float m[8], l[8];
  v8f O[8];
#pragma unroll
  for (int v = 0; v < 8; ++v) { m[v] = -1e30f; l[v] = 0.0f; }
#pragma unroll
  for (int c = 0; c < 8; ++c) { v8f z = {}; O[c] = z; }

  const float scale = 0.08838834764831845f;  // 1/sqrt(128)
  const int kend = ((q0 + 16 + 31) / 32) * 32;

  for (int kb = 0; kb < kend; kb += 32) {
    // --- scores: 16 queries x 32 keys (two 16x16 C tiles)
    v8f s0 = {}, s1 = {};
#pragma unroll
    for (int j = 0; j < 4; ++j) {
      const int kd = 32 * j + 8 * half;
      const unsigned short* kp0 = Kh + (size_t)(kb + lr) * HD + kd;
      const unsigned short* kp1 = Kh + (size_t)(kb + 16 + lr) * HD + kd;
      v16bf bk0 = make_frag(kp0, kp0 + 16);
      v16bf bk1 = make_frag(kp1, kp1 + 16);
      s0 = WMMA_BF16(qf[j], bk0, s0);
      s1 = WMMA_BF16(qf[j], bk1, s1);
    }

    // --- scale + causal mask + online softmax
    float p0[8], p1[8], alpha[8];
#pragma unroll
    for (int v = 0; v < 8; ++v) {
      const int row = q0 + v + 8 * half;
      p0[v] = (kb + lr      <= row) ? s0[v] * scale : -1e30f;
      p1[v] = (kb + 16 + lr <= row) ? s1[v] * scale : -1e30f;
    }
#pragma unroll
    for (int v = 0; v < 8; ++v) {
      float mx = fmaxf(p0[v], p1[v]);
      for (int off = 1; off < 16; off <<= 1)
        mx = fmaxf(mx, __shfl_xor(mx, off, 32));
      const float nm = fmaxf(m[v], mx);
      alpha[v] = __expf(m[v] - nm);
      p0[v] = __expf(p0[v] - nm);
      p1[v] = __expf(p1[v] - nm);
      float rs = p0[v] + p1[v];
      for (int off = 1; off < 16; off <<= 1)
        rs += __shfl_xor(rs, off, 32);
      l[v] = l[v] * alpha[v] + rs;
      m[v] = nm;
    }
#pragma unroll
    for (int c = 0; c < 8; ++c)
#pragma unroll
      for (int v = 0; v < 8; ++v) O[c][v] *= alpha[v];

    // --- re-layout P (C-frag: col per lane) -> A-frag (row per lane) via LDS
#pragma unroll
    for (int v = 0; v < 8; ++v) {
      const int row = v + 8 * half;
      Pl[row * 40 + lr]      = f2bf(p0[v]);
      Pl[row * 40 + 16 + lr] = f2bf(p1[v]);
    }
    asm volatile("s_wait_dscnt 0" ::: "memory");
    v16bf pf = make_frag(&Pl[lr * 40 + 8 * half], &Pl[lr * 40 + 16 + 8 * half]);

    // --- O += P @ V  (V transposed: contiguous keys per dim)
#pragma unroll
    for (int c = 0; c < 8; ++c) {
      const unsigned short* vp = Vh + (size_t)(c * 16 + lr) * SN + kb + 8 * half;
      v16bf vf = make_frag(vp, vp + 16);
      O[c] = WMMA_BF16(pf, vf, O[c]);
    }
  }

  // --- epilogue: normalize, write bf16 to [B,S,H]
#pragma unroll
  for (int v = 0; v < 8; ++v) {
    const int row = q0 + v + 8 * half;
    const float inv = 1.0f / l[v];
    unsigned short* dst = AO + ((size_t)(b * SN + row) * (NHQ * HD)) + (size_t)h * HD;
#pragma unroll
    for (int c = 0; c < 8; ++c)
      dst[c * 16 + lr] = f2bf(O[c][v] * inv);
  }
}

// ---------------------------------------------------------------- host
extern "C" void kernel_launch(void* const* d_in, const int* in_sizes, int n_in,
                              void* d_out, int out_size, void* d_ws, size_t ws_size,
                              hipStream_t stream) {
  const float* X    = (const float*)d_in[0];
  const float* cosp = (const float*)d_in[1];
  const float* sinp = (const float*)d_in[2];
  const float* Wq   = (const float*)d_in[3];
  const float* Wk   = (const float*)d_in[4];
  const float* Wv   = (const float*)d_in[5];
  const float* Wo   = (const float*)d_in[6];
  float* out = (float*)d_out;

  const int M  = BN * SN;       // 4096 rows
  const int H  = HDIM;          // 2048
  const int KV = KVH * HD;      // 512

  char* ws = (char*)d_ws;
  size_t off = 0;
  auto walloc = [&](size_t bytes) -> void* {
    void* p = ws + off;
    off += (bytes + 255) & ~(size_t)255;
    return p;
  };
  unsigned short* Xbf = (unsigned short*)walloc((size_t)M * H * 2);
  unsigned short* Wqb = (unsigned short*)walloc((size_t)H * H * 2);
  unsigned short* Wkb = (unsigned short*)walloc((size_t)KV * H * 2);
  unsigned short* Wvb = (unsigned short*)walloc((size_t)KV * H * 2);
  unsigned short* Wob = (unsigned short*)walloc((size_t)H * H * 2);
  float* Qf = (float*)walloc((size_t)M * H * 4);
  float* Kf = (float*)walloc((size_t)M * KV * 4);
  float* Vf = (float*)walloc((size_t)M * KV * 4);
  unsigned short* Qbf = (unsigned short*)walloc((size_t)M * H * 2);
  unsigned short* Kbf = (unsigned short*)walloc((size_t)M * KV * 2);
  unsigned short* Vtb = (unsigned short*)walloc((size_t)M * KV * 2);
  unsigned short* AO  = (unsigned short*)Qf;  // reuse dead Q f32 buffer

  const dim3 blk(256);
  // bf16 conversions
  cvt_f32_bf16<<<(M * H + 255) / 256, blk, 0, stream>>>(X, Xbf, M * H);
  cvt_f32_bf16<<<(H * H + 255) / 256, blk, 0, stream>>>(Wq, Wqb, H * H);
  cvt_f32_bf16<<<(KV * H + 255) / 256, blk, 0, stream>>>(Wk, Wkb, KV * H);
  cvt_f32_bf16<<<(KV * H + 255) / 256, blk, 0, stream>>>(Wv, Wvb, KV * H);
  cvt_f32_bf16<<<(H * H + 255) / 256, blk, 0, stream>>>(Wo, Wob, H * H);

  // QKV projections (WMMA GEMMs): block tile 128x256
  gemm_bf16_nt<<<dim3(H / 256, M / 128), blk, 0, stream>>>(Xbf, Wqb, Qf, M, H, H);
  gemm_bf16_nt<<<dim3(KV / 256, M / 128), blk, 0, stream>>>(Xbf, Wkb, Kf, M, KV, H);
  gemm_bf16_nt<<<dim3(KV / 256, M / 128), blk, 0, stream>>>(Xbf, Wvb, Vf, M, KV, H);

  // RoPE (fp32) + layout to bf16
  rope_transpose<<<(M * H + 255) / 256, blk, 0, stream>>>(Qf, cosp, sinp, Qbf, NHQ, 1, 0);
  rope_transpose<<<(M * KV + 255) / 256, blk, 0, stream>>>(Kf, cosp, sinp, Kbf, KVH, 1, 0);
  rope_transpose<<<(M * KV + 255) / 256, blk, 0, stream>>>(Vf, cosp, sinp, Vtb, KVH, 0, 1);

  // causal flash attention (WMMA)
  const int nwaves = BN * KVH * GROUPS * (SN / 16);   // 4096
  flash_attn<<<nwaves / 8, blk, 0, stream>>>(Qbf, Kbf, Vtb, AO);

  // output projection (WMMA GEMM) -> f32 result
  gemm_bf16_nt<<<dim3(H / 256, M / 128), blk, 0, stream>>>(AO, Wob, out, M, H, H);
}
